// LinearSelfAttention_18330920419514
// MI455X (gfx1250) — compile-verified, matrix operand order checked by reading, not run on hardware
//
#include <hip/hip_runtime.h>

// ---------------------------------------------------------------------------
// Linear self-attention, MI455X (gfx1250), wave32 + WMMA bf16 + TDM.
//
// Roofline: ~69 GFLOP of GEMM (x@Wqkv 51.5G, out@Wout 17.2G) vs ~130 MB of
// HBM traffic (~6 us @ 23.3 TB/s). bf16 WMMA (16x16x32) with f32 accumulate
// is the throughput path; weights (2 MB) and x_bf (32 MB) are L2-resident
// (192 MB L2). QKV GEMM stages its 64x512 bf16 weight panel into LDS once
// per workgroup via the Tensor Data Mover (tensor_load_to_lds +
// s_wait_tensorcnt), and software-pipelines the A-fragment global loads.
// ---------------------------------------------------------------------------

#define BATCH 4
#define NSEQ  8192
#define DIM   512
#define NH    8
#define HD    64
#define TD    1536   // 3*DIM

typedef __attribute__((ext_vector_type(16))) __bf16 v16bf;
typedef __attribute__((ext_vector_type(8)))  __bf16 v8bf;
typedef __attribute__((ext_vector_type(8)))  float  v8f;
typedef __attribute__((ext_vector_type(4)))  unsigned int v4u;
typedef __attribute__((ext_vector_type(8)))  int v8i;
typedef __attribute__((ext_vector_type(4)))  int v4i;

// Load a 16-element bf16 fragment for WMMA A/B operands from a row-major
// source. Per ISA 16-bit A/B layout: lanes 0-15 hold K = {k0..k0+7,
// k0+16..23}, lanes 16-31 hold K = {k0+8..15, k0+24..31}. Two 16B loads.
__device__ __forceinline__ v16bf frag_load(const __bf16* base, int ld, int row,
                                           int k0, int lane) {
  const __bf16* p = base + (size_t)row * ld + k0 + ((lane < 16) ? 0 : 8);
  v8bf lo = *(const v8bf*)(p);
  v8bf hi = *(const v8bf*)(p + 16);
  v16bf r;
#pragma unroll
  for (int i = 0; i < 8; ++i) { r[i] = lo[i]; r[i + 8] = hi[i]; }
  return r;
}

__device__ __forceinline__ v8f wmma_bf16(v16bf a, v16bf b, v8f c) {
  return __builtin_amdgcn_wmma_f32_16x16x32_bf16(
      /*neg_a=*/false, a, /*neg_b=*/false, b,
      /*c_mod=*/(short)0, c, /*reuse_a=*/false, /*reuse_b=*/false);
}

__device__ __forceinline__ float elu1(float x) {
  return x > 0.f ? x + 1.f : __expf(x);   // elu(x)+1
}

// ---------------------------------------------------------------------------
// Kernel 0a: x (f32) -> x_bf (bf16)
// ---------------------------------------------------------------------------
__global__ __launch_bounds__(256) void cvt_x_kernel(const float* __restrict__ x,
                                                    __bf16* __restrict__ xbf,
                                                    int n) {
  int i = blockIdx.x * 256 + threadIdx.x;
  if (i < n) xbf[i] = (__bf16)x[i];
}

// ---------------------------------------------------------------------------
// Kernel 0b: W [K][NC] f32 -> WT [NC][K] bf16 (so B fragments load contiguous)
// ---------------------------------------------------------------------------
__global__ __launch_bounds__(256) void cvt_wT_kernel(const float* __restrict__ w,
                                                     __bf16* __restrict__ wT,
                                                     int K, int NC) {
  int i = blockIdx.x * 256 + threadIdx.x;   // flat index of output [NC][K]
  if (i < K * NC) {
    int nn = i / K;
    int k  = i - nn * K;
    wT[i] = (__bf16)w[(size_t)k * NC + nn];
  }
}

// ---------------------------------------------------------------------------
// Kernel 1: qkv = x @ Wqkv  (bf16 WMMA, f32 acc), fused elu+1 epilogue,
// scatter into head-major q/k/v bf16 buffers [b][h][n][hd].
// Workgroup tile: 128(M) x 64(N), K=512. Wave w -> rows [row0+16w, +16).
// Weight panel (64 cols x 512 K bf16 = 64 KB) staged into LDS once per
// block via the Tensor Data Mover; A-fragments software-pipelined.
// ---------------------------------------------------------------------------
__global__ __launch_bounds__(256) void qkv_gemm_kernel(
    const __bf16* __restrict__ xbf, const __bf16* __restrict__ wT,
    __bf16* __restrict__ qf, __bf16* __restrict__ kf, __bf16* __restrict__ vf) {
  __shared__ __bf16 bpanel[64][DIM];   // 64 KB weight panel [col][k]

  const int lane = threadIdx.x & 31;
  const int l15  = lane & 15;
  const int sel8 = (lane < 16) ? 0 : 8;
  const int w    = threadIdx.x >> 5;
  const int mtb  = blockIdx.x / 24;          // 256 row blocks of 128
  const int cg   = blockIdx.x - mtb * 24;    // 24 col groups of 64
  const int row0 = mtb * 128 + w * 16;
  const int col0 = cg * 64;

  // --- TDM: DMA the 64x512 bf16 weight panel (contiguous rows) into LDS ---
  if (w == 0) {
    const unsigned long long ga =
        (unsigned long long)(uintptr_t)(wT + (size_t)col0 * DIM);
    const unsigned int lds_off = (unsigned int)(uintptr_t)(&bpanel[0][0]);
    // D# group 0: count=1 | lds_addr | global_addr[56:0] | type=2
    v4u g0;
    g0[0] = 1u;
    g0[1] = lds_off;
    g0[2] = (unsigned int)(ga & 0xFFFFFFFFu);
    g0[3] = (unsigned int)((ga >> 32) & 0x01FFFFFFu) | (2u << 30);
    // D# group 1: data_size=2B; tensor 512 x 1536; tile 512 x 64; stride 512
    v8i g1;
    g1[0] = (int)(1u << 16);                         // data_size=1 -> 2 bytes
    g1[1] = (int)((512u & 0xFFFFu) << 16);           // tensor_dim0[15:0]
    g1[2] = (int)(((512u >> 16) & 0xFFFFu) |         // tensor_dim0[31:16]
                  ((1536u & 0xFFFFu) << 16));        // tensor_dim1[15:0]
    g1[3] = (int)(((1536u >> 16) & 0xFFFFu) |        // tensor_dim1[31:16]
                  (512u << 16));                     // tile_dim0 = 512
    g1[4] = 64;                                      // tile_dim1=64, tile_dim2=0
    g1[5] = 512;                                     // tensor_dim0_stride lo
    g1[6] = 0;
    g1[7] = 0;
    v4i gz4 = {0, 0, 0, 0};                          // groups 2/3 unused (2D)
    v8i gz8 = {0, 0, 0, 0, 0, 0, 0, 0};              // extra group (clang-23 form)
    __builtin_amdgcn_tensor_load_to_lds(g0, g1, gz4, gz4, gz8, 0);
    __builtin_amdgcn_s_wait_tensorcnt((short)0);
  }
  __syncthreads();

  // --- main loop: A from global (pipelined), B from LDS panel ---
  v8f acc[4] = {};
  v16bf a_cur = frag_load(xbf, DIM, row0 + l15, 0, lane);
  for (int kk = 0; kk < DIM; kk += 32) {
    v16bf a_next;
    if (kk + 32 < DIM)
      a_next = frag_load(xbf, DIM, row0 + l15, kk + 32, lane);
#pragma unroll
    for (int j = 0; j < 4; ++j) {
      v16bf b = frag_load(&bpanel[0][0], DIM, j * 16 + l15, kk, lane);
      acc[j] = wmma_bf16(a_cur, b, acc[j]);
    }
    a_cur = a_next;
  }

#pragma unroll
  for (int j = 0; j < 4; ++j) {
    const int c    = col0 + j * 16 + l15;  // column in [0,1536)
    const int part = c >> 9;               // 0=q 1=k 2=v
    const int cc   = c & 511;
    const int h    = cc >> 6;
    const int e    = cc & 63;
    __bf16* dst = (part == 0) ? qf : (part == 1) ? kf : vf;
#pragma unroll
    for (int r = 0; r < 8; ++r) {
      const int g  = row0 + r + sel8;      // global M row
      const int bb = g >> 13;              // / NSEQ
      const int n  = g & (NSEQ - 1);
      float v = acc[j][r];
      if (part < 2) v = elu1(v);
      dst[(((size_t)bb * NH + h) * NSEQ + n) * HD + e] = (__bf16)v;
    }
  }
}

// ---------------------------------------------------------------------------
// Kernel 2: kv[b,h,d,e] = sum_n k[n,d] v[n,e] ; ksum[b,h,d] = sum_n k[n,d]
// Only ~2 GFLOP: VALU outer-product with LDS staging + f32 global atomics.
// ---------------------------------------------------------------------------
#define KV_SPLIT 16
__global__ __launch_bounds__(256) void kv_reduce_kernel(
    const __bf16* __restrict__ kf, const __bf16* __restrict__ vf,
    float* __restrict__ kvg, float* __restrict__ ksumg) {
  __shared__ float lk[32][HD + 1];
  __shared__ float lv[32][HD + 1];
  const int bh = blockIdx.x / KV_SPLIT;
  const int sp = blockIdx.x - bh * KV_SPLIT;
  const int t  = threadIdx.x;
  const int d  = t >> 2;
  const int e0 = (t & 3) << 4;

  float acc[16] = {};
  float ks = 0.f;
  const size_t base = ((size_t)bh * NSEQ + (size_t)sp * (NSEQ / KV_SPLIT)) * HD;

  for (int c = 0; c < NSEQ / KV_SPLIT; c += 32) {
    for (int i = t; i < 32 * HD; i += 256) {
      const int r = i >> 6, col = i & 63;
      lk[r][col] = (float)kf[base + (size_t)(c + r) * HD + col];
      lv[r][col] = (float)vf[base + (size_t)(c + r) * HD + col];
    }
    __syncthreads();
#pragma unroll 4
    for (int r = 0; r < 32; ++r) {
      const float kd = lk[r][d];
      if ((t & 3) == 0) ks += kd;
#pragma unroll
      for (int j = 0; j < 16; ++j) acc[j] += kd * lv[r][e0 + j];
    }
    __syncthreads();
  }

  float* kvdst = kvg + (size_t)bh * HD * HD + d * HD + e0;
#pragma unroll
  for (int j = 0; j < 16; ++j) atomicAdd(kvdst + j, acc[j]);
  if ((t & 3) == 0) atomicAdd(ksumg + bh * HD + d, ks);
}

// ---------------------------------------------------------------------------
// Kernel 3: per 16-row block:
//   z[m,h] = 1/(q[m,:].ksum[h] + 1e-6)
//   ao[m, h*64+e] = (q_h @ kv_h)[m,e] * z[m,h]        (WMMA, K=64)
//   out[m, :] = ao[m,:] @ Wout + bout                 (WMMA, K=512)
// ---------------------------------------------------------------------------
__global__ __launch_bounds__(256) void attn_out_kernel(
    const __bf16* __restrict__ qf, const float* __restrict__ kvg,
    const float* __restrict__ ksumg, const __bf16* __restrict__ woutT,
    const float* __restrict__ bout, float* __restrict__ out) {
  __shared__ __bf16 kvT[NH][HD][HD];   // [h][e][d]  64 KB
  __shared__ float  ksl[NH][HD];
  __shared__ float  zl[16][NH];
  __shared__ __bf16 ao[16][DIM];       // 16 KB

  const int t    = threadIdx.x;
  const int lane = t & 31;
  const int l15  = lane & 15;
  const int sel8 = (lane < 16) ? 0 : 8;
  const int w    = t >> 5;                 // wave id == head id for stage 1
  const int bb   = blockIdx.x >> 9;        // / (NSEQ/16)
  const int row0 = (blockIdx.x & 511) * 16;

  // Stage kv (transposed, bf16) + ksum into LDS
  const size_t kvbase = (size_t)bb * NH * HD * HD;
  for (int i = t; i < NH * HD * HD; i += 256) {
    const int h = i >> 12;
    const int d = (i >> 6) & 63;
    const int e = i & 63;
    kvT[h][e][d] = (__bf16)kvg[kvbase + i];
  }
  for (int i = t; i < NH * HD; i += 256)
    ksl[i >> 6][i & 63] = ksumg[(size_t)bb * NH * HD + i];
  __syncthreads();

  // z per (row, head)
  if (t < 128) {
    const int m = t >> 3;
    const int h = t & 7;
    const __bf16* qrow = qf + (((size_t)bb * NH + h) * NSEQ + row0 + m) * HD;
    float s = 0.f;
#pragma unroll
    for (int i = 0; i < HD; ++i) s += (float)qrow[i] * ksl[h][i];
    zl[m][h] = 1.f / (s + 1e-6f);
  }
  __syncthreads();

  // Stage 1: wave w computes head w: (16 x 64) = q_tile(16x64) @ kv(64x64)
  {
    const __bf16* qbase = qf + (((size_t)bb * NH + w) * NSEQ) * HD;
    v8f acc[4] = {};
    for (int kk = 0; kk < HD; kk += 32) {
      v16bf a = frag_load(qbase, HD, row0 + l15, kk, lane);
#pragma unroll
      for (int j = 0; j < 4; ++j) {
        v16bf b = frag_load(&kvT[w][0][0], HD, j * 16 + l15, kk, lane);
        acc[j] = wmma_bf16(a, b, acc[j]);
      }
    }
#pragma unroll
    for (int j = 0; j < 4; ++j) {
#pragma unroll
      for (int r = 0; r < 8; ++r) {
        const int m = r + sel8;
        ao[m][w * HD + j * 16 + l15] = (__bf16)(acc[j][r] * zl[m][w]);
      }
    }
  }
  __syncthreads();

  // Stage 2: out(16 x 512) = ao(16x512) @ WoutT + bout
  {
    v8f acc[4] = {};
    for (int kk = 0; kk < DIM; kk += 32) {
      v16bf a = frag_load(&ao[0][0], DIM, l15, kk, lane);
#pragma unroll
      for (int j = 0; j < 4; ++j) {
        const int col0 = (j * 8 + w) * 16;
        v16bf b = frag_load(woutT, DIM, col0 + l15, kk, lane);
        acc[j] = wmma_bf16(a, b, acc[j]);
      }
    }
#pragma unroll
    for (int j = 0; j < 4; ++j) {
      const int col  = (j * 8 + w) * 16 + l15;
      const float bias = bout[col];
#pragma unroll
      for (int r = 0; r < 8; ++r) {
        const int m = r + sel8;
        out[((size_t)bb * NSEQ + row0 + m) * DIM + col] = acc[j][r] + bias;
      }
    }
  }
}

// ---------------------------------------------------------------------------
// Host launcher. Workspace layout (bytes, all 256-aligned):
//   x_bf   [B*N*D] bf16             @ 0          (32 MB)
//   wqkvT  [1536][512] bf16         @ 33554432   (1.5 MB)
//   woutT  [512][512] bf16          @ 35127296   (0.5 MB)
//   qf/kf/vf [B][H][N][HD] bf16     @ 35651584.. (3 x 32 MB)
//   kv     [B*H][64][64] f32        @ 136314880  (512 KB)
//   ksum   [B*H][64] f32            @ 136839168  (8 KB)
// ---------------------------------------------------------------------------
extern "C" void kernel_launch(void* const* d_in, const int* in_sizes, int n_in,
                              void* d_out, int out_size, void* d_ws, size_t ws_size,
                              hipStream_t stream) {
  const float* x    = (const float*)d_in[0];
  const float* Wqkv = (const float*)d_in[1];
  const float* Wout = (const float*)d_in[2];
  const float* bout = (const float*)d_in[3];
  float* out = (float*)d_out;

  char* ws = (char*)d_ws;
  const size_t XBF   = 0;
  const size_t WQKVT = XBF + (size_t)BATCH * NSEQ * DIM * 2;      // 33554432
  const size_t WOUTT = WQKVT + (size_t)DIM * TD * 2;              // +1572864
  const size_t QF    = WOUTT + (size_t)DIM * DIM * 2;             // +524288
  const size_t KF    = QF + (size_t)BATCH * NSEQ * DIM * 2;
  const size_t VF    = KF + (size_t)BATCH * NSEQ * DIM * 2;
  const size_t KV    = VF + (size_t)BATCH * NSEQ * DIM * 2;
  const size_t KSUM  = KV + (size_t)BATCH * NH * HD * HD * 4;

  __bf16* xbf   = (__bf16*)(ws + XBF);
  __bf16* wqkvT = (__bf16*)(ws + WQKVT);
  __bf16* woutT = (__bf16*)(ws + WOUTT);
  __bf16* qf    = (__bf16*)(ws + QF);
  __bf16* kf    = (__bf16*)(ws + KF);
  __bf16* vf    = (__bf16*)(ws + VF);
  float*  kvg   = (float*)(ws + KV);
  float*  ksumg = (float*)(ws + KSUM);

  // zero kv + ksum accumulators (contiguous)
  (void)hipMemsetAsync(kvg, 0,
                       (size_t)BATCH * NH * HD * HD * 4 + (size_t)BATCH * NH * HD * 4,
                       stream);

  const int nx = BATCH * NSEQ * DIM;
  cvt_x_kernel<<<(nx + 255) / 256, 256, 0, stream>>>(x, xbf, nx);
  cvt_wT_kernel<<<(DIM * TD + 255) / 256, 256, 0, stream>>>(Wqkv, wqkvT, DIM, TD);
  cvt_wT_kernel<<<(DIM * DIM + 255) / 256, 256, 0, stream>>>(Wout, woutT, DIM, DIM);

  // 256 row blocks (128 rows) * 24 col groups, 8 waves per block
  qkv_gemm_kernel<<<256 * 24, 256, 0, stream>>>(xbf, wqkvT, qf, kf, vf);

  kv_reduce_kernel<<<BATCH * NH * KV_SPLIT, 256, 0, stream>>>(kf, vf, kvg, ksumg);

  attn_out_kernel<<<BATCH * (NSEQ / 16), 256, 0, stream>>>(qf, kvg, ksumg, woutT,
                                                           bout, out);
}
